// SPDGeoAltMoE_48009144435314
// MI455X (gfx1250) — compile-verified, matrix operand order checked by reading, not run on hardware
//
#include <hip/hip_runtime.h>
#include <math.h>

// ---------- problem constants ----------
#define BB_   256      // batch
#define NN_   576      // patches
#define FF_   384      // feat dim in
#define DD_   256      // part dim
#define KP_   8        // n parts
#define NE_   4        // n experts
#define MM_   (BB_*NN_)   // 147456 rows

// d_out layout (floats, reference return order)
#define OFF_PART   0
#define OFF_ASSIGN 524288
#define OFF_SAL    1703936
#define OFF_FEAT   1705984
#define OFF_GATE   39454720

typedef __bf16 bf16_t;
typedef __attribute__((ext_vector_type(16))) __bf16 v16bf;
typedef __attribute__((ext_vector_type(8)))  float  v8f;
typedef __attribute__((ext_vector_type(4)))  unsigned int u32x4;
typedef __attribute__((ext_vector_type(8)))  int i32x8;
typedef __attribute__((ext_vector_type(4)))  int i32x4;

__device__ __forceinline__ float gelu_f(float x) {
    return 0.5f * x * (1.0f + erff(x * 0.70710678118654752f));
}

// ---------------------------------------------------------------------
// TDM: build a D# (ISA 8.3/8.4) and issue tensor_load_to_lds.
// 2-D tile: tile_d0 contiguous elements per row, tile_d1 rows,
// row stride = stride0 elements.  dsz_log: 0=1B,1=2B,2=4B,3=8B.
// Tracked on TENSORcnt (wave-level, EXEC-ignored).
// clang-23 toolchain: 6-arg builtin (g0, g1, g2, g3, g4, cpol).
// ---------------------------------------------------------------------
__device__ __forceinline__ void tdm_load_2d(
    unsigned lds_off, const void* gptr,
    unsigned tile_d0, unsigned tile_d1,
    unsigned tensor_d0, unsigned tensor_d1,
    unsigned stride0, unsigned dsz_log)
{
    const unsigned long long ga = (unsigned long long)gptr;
    u32x4 g0;
    g0[0] = 1u;                                        // count=1, user D#
    g0[1] = lds_off;                                   // lds_addr (bytes)
    g0[2] = (unsigned)(ga & 0xffffffffu);              // global_addr[31:0]
    g0[3] = (unsigned)((ga >> 32) & 0x1ffffffu)        // global_addr[56:32]
          | (2u << 30);                                // type=2 ("image")
    i32x8 g1;
    g1[0] = (int)(dsz_log << 16);                      // wg_mask=0, data_size
    g1[1] = (int)(tensor_d0 << 16);                    // tensor_dim0[15:0]
    g1[2] = (int)((tensor_d0 >> 16) & 0xffffu) | (int)(tensor_d1 << 16);
    g1[3] = (int)((tensor_d1 >> 16) & 0xffffu) | (int)(tile_d0 << 16);
    g1[4] = (int)(tile_d1 & 0xffffu);                  // tile_dim1, tile_dim2=0
    g1[5] = (int)stride0;                              // dim0_stride[31:0]
    g1[6] = 0;                                         // stride hi bits
    g1[7] = 0;
    i32x4 z4; z4[0] = 0; z4[1] = 0; z4[2] = 0; z4[3] = 0;   // groups 2/3 (<=2D)
    i32x8 z8;
    z8[0] = 0; z8[1] = 0; z8[2] = 0; z8[3] = 0;
    z8[4] = 0; z8[5] = 0; z8[6] = 0; z8[7] = 0;
    __builtin_amdgcn_tensor_load_to_lds(g0, g1, z4, z4, z8, 0);
}

// =====================================================================
// K0: pre-convert w_proj [384,256] f32 -> Wt [256,384] bf16 (k-contig)
// =====================================================================
__global__ __launch_bounds__(256) void k0_wt(
    const float* __restrict__ wp, bf16_t* __restrict__ wt)
{
    const int k = blockIdx.x;      // 0..383
    const int c = threadIdx.x;     // 0..255
    wt[(size_t)c * FF_ + k] = (__bf16)wp[(size_t)k * DD_ + c];
}

// =====================================================================
// K1: feat = GELU(LN(patch @ Wp + b)).  32 rows x 256 cols per block.
// W tiles arrive via double-buffered TDM (TENSORcnt); A tile staged
// manually (needs f32->bf16).  LN stats from WMMA accumulators via
// shfl + LDS row atomics; outputs written straight from registers.
// =====================================================================
__global__ __launch_bounds__(256) void k1_proj(
    const float* __restrict__ patch, const bf16_t* __restrict__ wt,
    const float* __restrict__ bp, const float* __restrict__ lng,
    const float* __restrict__ lnb,
    float* __restrict__ featOut, bf16_t* __restrict__ featb,
    float* __restrict__ stats)
{
    __shared__ __attribute__((aligned(32))) __bf16 As[32][32];
    __shared__ __attribute__((aligned(64))) __bf16 Ws[2][256][32];
    __shared__ float rowsum[32], rowsumsq[32];
    __shared__ float colsum[256];

    const int t  = threadIdx.x;
    const int m0 = blockIdx.x * 32;
    const int w  = t >> 5;
    const int l  = t & 31;
    const int rt = w & 1;          // row tile 0..1
    const int cg = w >> 1;         // col group 0..3
    const int rl = l & 15;
    const int hi = l >> 4;

    colsum[t] = 0.0f;
    if (t < 32) { rowsum[t] = 0.0f; rowsumsq[t] = 0.0f; }

    v8f acc[4];
#pragma unroll
    for (int ct = 0; ct < 4; ct++)
#pragma unroll
        for (int j = 0; j < 8; j++) acc[ct][j] = 0.0f;

    const int lrow = t >> 3;           // A staging map
    const int lkq  = (t & 7) * 4;
    const int arow = rt * 16 + rl;
    const int koff = hi * 16;

    // prologue: DMA first W tile into buffer 0
    if (w == 0)
        tdm_load_2d((unsigned)(size_t)&Ws[0][0][0], wt,
                    32, 256, FF_, 256, FF_, 1);

    for (int i = 0; i < 12; i++) {
        const int kk = i * 32;
        // stage A tile (32x32) f32 -> bf16
        {
            const float4 v = *(const float4*)(patch + (size_t)(m0 + lrow) * FF_ + kk + lkq);
            As[lrow][lkq + 0] = (__bf16)v.x;
            As[lrow][lkq + 1] = (__bf16)v.y;
            As[lrow][lkq + 2] = (__bf16)v.z;
            As[lrow][lkq + 3] = (__bf16)v.w;
            __builtin_prefetch(patch + (size_t)(m0 + lrow) * FF_ + kk + 32 + lkq, 0, 1);
        }
        // overlap: issue next W tile, then wait only for the older one
        if (w == 0) {
            if (i + 1 < 12) {
                tdm_load_2d((unsigned)(size_t)&Ws[(i + 1) & 1][0][0], wt + kk + 32,
                            32, 256, FF_, 256, FF_, 1);
                __builtin_amdgcn_s_wait_tensorcnt(1);
            } else {
                __builtin_amdgcn_s_wait_tensorcnt(0);
            }
        }
        __syncthreads();

        const v16bf afrag = *(const v16bf*)&As[arow][koff];
        const __bf16 (*wsb)[32] = Ws[i & 1];
#pragma unroll
        for (int ct = 0; ct < 4; ct++) {
            const int col = cg * 64 + ct * 16 + rl;
            const v16bf bfrag = *(const v16bf*)&wsb[col][koff];
            acc[ct] = __builtin_amdgcn_wmma_f32_16x16x32_bf16(
                false, afrag, false, bfrag, (short)0, acc[ct], false, false);
        }
        __syncthreads();
    }

    // bias + partial row sums (per half-wave: 16 lanes share a row)
    float biasv[4], lngv[4], lnbv[4];
#pragma unroll
    for (int ct = 0; ct < 4; ct++) {
        const int col = cg * 64 + ct * 16 + rl;
        biasv[ct] = bp[col];
        lngv[ct]  = lng[col];
        lnbv[ct]  = lnb[col];
    }
#pragma unroll
    for (int j = 0; j < 8; j++) {
        float s = 0.0f, ss = 0.0f;
#pragma unroll
        for (int ct = 0; ct < 4; ct++) {
            const float v = acc[ct][j] + biasv[ct];
            acc[ct][j] = v;
            s += v; ss += v * v;
        }
#pragma unroll
        for (int off = 8; off > 0; off >>= 1) {
            s  += __shfl_xor(s,  off, 32);
            ss += __shfl_xor(ss, off, 32);
        }
        if (rl == 0) {
            const int row = rt * 16 + j + hi * 8;
            atomicAdd(&rowsum[row], s);
            atomicAdd(&rowsumsq[row], ss);
        }
    }
    __syncthreads();

    // LN + GELU + stores, straight from registers
#pragma unroll
    for (int j = 0; j < 8; j++) {
        const int row = rt * 16 + j + hi * 8;
        const float mu  = rowsum[row] * (1.0f / 256.0f);
        const float var = rowsumsq[row] * (1.0f / 256.0f) - mu * mu;
        const float rs  = rsqrtf(var + 1e-5f);
        const size_t gm = (size_t)(m0 + row) * DD_;
#pragma unroll
        for (int ct = 0; ct < 4; ct++) {
            const int col = cg * 64 + ct * 16 + rl;
            const float xv = (acc[ct][j] - mu) * rs * lngv[ct] + lnbv[ct];
            const float y  = gelu_f(xv);
            featOut[gm + col] = y;
            featb[gm + col]   = (__bf16)y;
            atomicAdd(&colsum[col], y);
        }
    }
    __syncthreads();
    atomicAdd(&stats[(m0 / NN_) * DD_ + t], colsum[t]);
}

// =====================================================================
// K2: feat_stats finalize, router MLP, gate softmax, blended+l2norm
//     prototypes (bf16).  One block per batch element.
// =====================================================================
__global__ __launch_bounds__(256) void k2_router(
    const float* __restrict__ stats, const int* __restrict__ alt_idx,
    const int* __restrict__ epoch,
    const float* __restrict__ r_alt_emb, const float* __restrict__ r_wf,
    const float* __restrict__ r_bf, const float* __restrict__ r_w1,
    const float* __restrict__ r_b1, const float* __restrict__ r_w2,
    const float* __restrict__ r_b2, const float* __restrict__ protos,
    float* __restrict__ gateOut, bf16_t* __restrict__ protn)
{
    __shared__ float fs[256];
    __shared__ float hin[128];
    __shared__ float hv[64];
    __shared__ float lg[4], gw[4];
    __shared__ float knorm[8];
    __shared__ float parr[2048];

    const int t = threadIdx.x;
    const int b = blockIdx.x;

    fs[t] = stats[b * DD_ + t] * (1.0f / (float)NN_);
    if (t < 8) knorm[t] = 0.0f;
    __syncthreads();

    if (t < 64) {
        float s = r_bf[t];
        for (int i = 0; i < 256; i++) s += fs[i] * r_wf[i * 64 + t];
        hin[t]      = fmaxf(s, 0.0f);
        hin[64 + t] = r_alt_emb[alt_idx[b] * 64 + t];
    }
    __syncthreads();
    if (t < 64) {
        float s = r_b1[t];
        for (int i = 0; i < 128; i++) s += hin[i] * r_w1[i * 64 + t];
        hv[t] = fmaxf(s, 0.0f);
    }
    __syncthreads();
    if (t < 4) {
        const float ep   = (float)epoch[0];
        const float warm = fminf(ep * 0.125f, 1.0f);
        const float temp = 2.0f * (1.0f - warm) + 0.5f * warm;
        float s = r_b2[t];
        for (int i = 0; i < 64; i++) s += hv[i] * r_w2[i * 4 + t];
        lg[t] = s / temp;
    }
    __syncthreads();
    if (t == 0) {
        const float mx = fmaxf(fmaxf(lg[0], lg[1]), fmaxf(lg[2], lg[3]));
        float e[4], se = 0.0f;
        for (int i = 0; i < 4; i++) { e[i] = expf(lg[i] - mx); se += e[i]; }
        for (int i = 0; i < 4; i++) gw[i] = e[i] / se;
    }
    __syncthreads();
    if (t < 4) gateOut[b * NE_ + t] = gw[t];

    for (int idx = t; idx < KP_ * DD_; idx += 256) {
        const int k = idx >> 8;
        float s = 0.0f;
        for (int e = 0; e < NE_; e++)
            s += gw[e] * protos[((size_t)e * KP_ + k) * DD_ + (idx & 255)];
        parr[idx] = s;
        atomicAdd(&knorm[k], s * s);
    }
    __syncthreads();
    for (int idx = t; idx < KP_ * DD_; idx += 256) {
        const int k = idx >> 8;
        const float inv = 1.0f / fmaxf(sqrtf(knorm[k]), 1e-12f);
        protn[(size_t)b * (KP_ * DD_) + idx] = (__bf16)(parr[idx] * inv);
    }
}

// =====================================================================
// K3: sim = l2norm(feat)·protn / 0.07 via WMMA (8 parts padded to 16
//     cols), softmax over parts, write assign (f32) + transposed bf16,
//     accumulate mass.  Block 128 thr = 4 waves x 16 rows = 64 rows.
// =====================================================================
__global__ __launch_bounds__(128) void k3_assign(
    const bf16_t* __restrict__ featb, const bf16_t* __restrict__ protn,
    float* __restrict__ assignOut, bf16_t* __restrict__ at,
    float* __restrict__ mass)
{
    __shared__ float Cs[4][16][16];
    __shared__ float bmass[8];

    const int t = threadIdx.x;
    const int w = t >> 5;
    const int l = t & 31;
    if (t < 8) bmass[t] = 0.0f;

    const int m0   = blockIdx.x * 64 + w * 16;   // 16 | 576 -> same batch
    const int b    = m0 / NN_;
    const int rloc = l & 15;
    const int hi   = l >> 4;
    const int kp   = rloc;

    v8f acc;
#pragma unroll
    for (int j = 0; j < 8; j++) acc[j] = 0.0f;

    const bf16_t* arow = featb + (size_t)(m0 + rloc) * DD_;
    const bf16_t* brow = protn + ((size_t)b * KP_ + kp) * DD_;  // valid kp<8
    float ssq = 0.0f;

#pragma unroll
    for (int step = 0; step < 8; step++) {
        const int doff = hi * 16 + step * 32;
        const v16bf afrag = *(const v16bf*)(arow + doff);
#pragma unroll
        for (int i = 0; i < 16; i++) { const float v = (float)afrag[i]; ssq += v * v; }
        v16bf bfrag;
#pragma unroll
        for (int i = 0; i < 16; i++) bfrag[i] = (__bf16)0.0f;
        if (kp < KP_) bfrag = *(const v16bf*)(brow + doff);
        acc = __builtin_amdgcn_wmma_f32_16x16x32_bf16(
            false, afrag, false, bfrag, (short)0, acc, false, false);
    }

    const float ssqT = ssq + __shfl_xor(ssq, 16, 32);
#pragma unroll
    for (int j = 0; j < 8; j++) Cs[w][j + hi * 8][rloc] = acc[j];
    __syncthreads();

    if (l < 16) {
        const int row = m0 + l;
        const float scale = (1.0f / 0.07f) / fmaxf(sqrtf(ssqT), 1e-12f);
        float sv[8], mx = -1e30f;
#pragma unroll
        for (int k = 0; k < KP_; k++) { sv[k] = Cs[w][l][k] * scale; mx = fmaxf(mx, sv[k]); }
        float se = 0.0f;
#pragma unroll
        for (int k = 0; k < KP_; k++) { sv[k] = expf(sv[k] - mx); se += sv[k]; }
        const float inv = 1.0f / se;
#pragma unroll
        for (int k = 0; k < KP_; k++) {
            const float a = sv[k] * inv;
            assignOut[(size_t)row * KP_ + k] = a;
            at[((size_t)b * KP_ + k) * NN_ + (row - b * NN_)] = (__bf16)a;
            atomicAdd(&bmass[k], a);
        }
    }
    __syncthreads();
    if (t < 8) atomicAdd(&mass[b * KP_ + t], bmass[t]);
}

// =====================================================================
// K4: part_feat_raw = (assign^T @ feat) / mass.  One block per batch.
//     feat rows arrive via double-buffered 1-D TDM transfers.
// =====================================================================
__global__ __launch_bounds__(256) void k4_part(
    const bf16_t* __restrict__ featb, const bf16_t* __restrict__ at,
    const float* __restrict__ mass, float* __restrict__ partOut)
{
    __shared__ __attribute__((aligned(64))) __bf16 fbuf[2][8][256];
    __shared__ float abuf[8][8];

    const int t  = threadIdx.x;
    const int b  = blockIdx.x;
    const int kp = t >> 5;
    const int d0 = (t & 31) * 8;
    const bf16_t* fb = featb + (size_t)b * NN_ * DD_;

    float acc[8];
#pragma unroll
    for (int i = 0; i < 8; i++) acc[i] = 0.0f;

    if (t < 32)   // wave 0 drives the DMA pipeline
        tdm_load_2d((unsigned)(size_t)&fbuf[0][0][0], fb,
                    8 * DD_, 0, 8 * DD_, 1, 8 * DD_, 1);

    for (int i = 0; i < NN_ / 8; i++) {
        const int n = i * 8;
        if (t < 64) {
            const int nn = t >> 3, k = t & 7;
            abuf[nn][k] = (float)at[((size_t)b * KP_ + k) * NN_ + n + nn];
        }
        if (t < 32) {
            if (i + 1 < NN_ / 8) {
                tdm_load_2d((unsigned)(size_t)&fbuf[(i + 1) & 1][0][0],
                            fb + (size_t)(n + 8) * DD_,
                            8 * DD_, 0, 8 * DD_, 1, 8 * DD_, 1);
                __builtin_amdgcn_s_wait_tensorcnt(1);
            } else {
                __builtin_amdgcn_s_wait_tensorcnt(0);
            }
        }
        __syncthreads();
        const __bf16 (*fc)[256] = fbuf[i & 1];
#pragma unroll
        for (int j = 0; j < 8; j++) {
            const float aw = abuf[j][kp];
#pragma unroll
            for (int ii = 0; ii < 8; ii++) acc[ii] += aw * (float)fc[j][d0 + ii];
        }
        __syncthreads();
    }
    const float invm = 1.0f / fmaxf(mass[b * KP_ + kp], 1e-6f);
#pragma unroll
    for (int i = 0; i < 8; i++)
        partOut[((size_t)b * KP_ + kp) * DD_ + d0 + i] = acc[i] * invm;
}

// =====================================================================
// K5: refine residual MLP + salience head.  One block per (b,k) row.
// =====================================================================
__global__ __launch_bounds__(256) void k5_refine(
    float* __restrict__ part, float* __restrict__ salOut,
    const float* __restrict__ lng, const float* __restrict__ lnb,
    const float* __restrict__ w1, const float* __restrict__ b1,
    const float* __restrict__ w2, const float* __restrict__ b2,
    const float* __restrict__ sw1, const float* __restrict__ sb1,
    const float* __restrict__ sw2, const float* __restrict__ sb2)
{
    __shared__ float row[256];
    __shared__ float rln[256];
    __shared__ float h1[512];
    __shared__ float red[256];
    __shared__ float s1[64];

    const int t = threadIdx.x;
    const size_t r = blockIdx.x;    // 0..B*KP-1

    const float v = part[r * DD_ + t];
    row[t] = v;
    red[t] = v;
    __syncthreads();
    for (int off = 128; off > 0; off >>= 1) { if (t < off) red[t] += red[t + off]; __syncthreads(); }
    const float mu = red[0] * (1.0f / 256.0f);
    __syncthreads();
    red[t] = (v - mu) * (v - mu);
    __syncthreads();
    for (int off = 128; off > 0; off >>= 1) { if (t < off) red[t] += red[t + off]; __syncthreads(); }
    const float rs = rsqrtf(red[0] * (1.0f / 256.0f) + 1e-5f);
    rln[t] = (v - mu) * rs * lng[t] + lnb[t];
    __syncthreads();

    for (int o = t; o < 512; o += 256) {
        float s = b1[o];
        for (int i = 0; i < 256; i++) s += rln[i] * w1[i * 512 + o];
        h1[o] = gelu_f(s);
    }
    __syncthreads();
    {
        float s = b2[t];
        for (int i = 0; i < 512; i++) s += h1[i] * w2[i * 256 + t];
        const float np = v + s;
        part[r * DD_ + t] = np;
        row[t] = np;
    }
    __syncthreads();
    if (t < 64) {
        float s = sb1[t];
        for (int i = 0; i < 256; i++) s += row[i] * sw1[i * 64 + t];
        s1[t] = gelu_f(s);
    }
    __syncthreads();
    if (t == 0) {
        float s = sb2[0];
        for (int i = 0; i < 64; i++) s += s1[i] * sw2[i];
        salOut[r] = 1.0f / (1.0f + expf(-s));
    }
}

// =====================================================================
extern "C" void kernel_launch(void* const* d_in, const int* in_sizes, int n_in,
                              void* d_out, int out_size, void* d_ws, size_t ws_size,
                              hipStream_t stream) {
    const float* patch   = (const float*)d_in[0];
    const int*   alt     = (const int*)d_in[1];
    const int*   epoch   = (const int*)d_in[2];
    const float* w_proj  = (const float*)d_in[3];
    const float* b_proj  = (const float*)d_in[4];
    const float* ln1_g   = (const float*)d_in[5];
    const float* ln1_b   = (const float*)d_in[6];
    const float* protos  = (const float*)d_in[7];
    const float* r_alt   = (const float*)d_in[8];
    const float* r_wf    = (const float*)d_in[9];
    const float* r_bf    = (const float*)d_in[10];
    const float* r_w1    = (const float*)d_in[11];
    const float* r_b1    = (const float*)d_in[12];
    const float* r_w2    = (const float*)d_in[13];
    const float* r_b2    = (const float*)d_in[14];
    const float* ref_lng = (const float*)d_in[15];
    const float* ref_lnb = (const float*)d_in[16];
    const float* ref_w1  = (const float*)d_in[17];
    const float* ref_b1  = (const float*)d_in[18];
    const float* ref_w2  = (const float*)d_in[19];
    const float* ref_b2  = (const float*)d_in[20];
    const float* sal_w1  = (const float*)d_in[21];
    const float* sal_b1  = (const float*)d_in[22];
    const float* sal_w2  = (const float*)d_in[23];
    const float* sal_b2  = (const float*)d_in[24];

    float* out    = (float*)d_out;
    float* part   = out + OFF_PART;
    float* assign = out + OFF_ASSIGN;
    float* sal    = out + OFF_SAL;
    float* feat   = out + OFF_FEAT;
    float* gate   = out + OFF_GATE;

    float*  stats = (float*)d_ws;                       // B*D
    float*  mass  = stats + BB_ * DD_;                  // B*KP
    bf16_t* protn = (bf16_t*)(mass + BB_ * KP_);        // B*KP*D bf16
    bf16_t* at    = protn + (size_t)BB_ * KP_ * DD_;    // B*KP*N bf16
    bf16_t* featb = at + (size_t)BB_ * KP_ * NN_;       // B*N*D bf16
    bf16_t* wt    = featb + (size_t)BB_ * NN_ * DD_;    // 256*384 bf16

    (void)hipMemsetAsync(stats, 0, (BB_ * DD_ + BB_ * KP_) * sizeof(float), stream);

    k0_wt<<<FF_, 256, 0, stream>>>(w_proj, wt);
    k1_proj<<<MM_ / 32, 256, 0, stream>>>(patch, wt, b_proj, ln1_g, ln1_b,
                                          feat, featb, stats);
    k2_router<<<BB_, 256, 0, stream>>>(stats, alt, epoch, r_alt, r_wf, r_bf,
                                       r_w1, r_b1, r_w2, r_b2, protos,
                                       gate, protn);
    k3_assign<<<MM_ / 64, 128, 0, stream>>>(featb, protn, assign, at, mass);
    k4_part<<<BB_, 256, 0, stream>>>(featb, at, mass, part);
    k5_refine<<<BB_ * KP_, 256, 0, stream>>>(part, sal, ref_lng, ref_lnb,
                                             ref_w1, ref_b1, ref_w2, ref_b2,
                                             sal_w1, sal_b1, sal_w2, sal_b2);
}
